// BindNode23GCN_62715112456262
// MI455X (gfx1250) — matrix operand order repaired and checked by reading, not compiled, not run on hardware
//
#include <hip/hip_runtime.h>
#include <hip/hip_bf16.h>

namespace {
constexpr int kNodes    = 70000;
constexpr int kEdges    = 2000000;
constexpr int kEdgesTot = kEdges + kNodes;   // self-loops appended
constexpr int kInCh     = 1044;
constexpr int kKMain    = (kInCh / 32) * 32; // 1024: unguarded K range
constexpr int kFeat     = 128;
constexpr int kOut      = 3;
constexpr int kBStride  = 40;                // bf16 k-stride per column (80 B)
}

typedef __attribute__((ext_vector_type(16))) __bf16 v16bf;
typedef __attribute__((ext_vector_type(8)))  float  v8f;

struct BfPair { uint4 lo, hi; };

// ---------------------------------------------------------------------------
// degree / norm helpers
// ---------------------------------------------------------------------------
__global__ void k_init_deg(float* deg, int n) {
  int i = blockIdx.x * blockDim.x + threadIdx.x;
  if (i < n) deg[i] = 1.0f;               // self-loop contributes 1 per node
}

__global__ void k_accum_deg(const long long* __restrict__ edges, float* deg, int nE) {
  int e = blockIdx.x * blockDim.x + threadIdx.x;
  if (e < nE) {
    int d = (int)edges[(size_t)nE + e];   // edges[1][e] = dst
    atomicAdd(&deg[d], 1.0f);
  }
}

__global__ void k_dinv(const float* __restrict__ deg, float* dinv, int n) {
  int i = blockIdx.x * blockDim.x + threadIdx.x;
  if (i < n) {
    float d = deg[i];
    dinv[i] = (d > 0.0f) ? rsqrtf(d) : 0.0f;
  }
}

__global__ void k_zero_f32(float* p, int n) {
  int i = blockIdx.x * blockDim.x + threadIdx.x;
  if (i < n) p[i] = 0.0f;
}

// ---------------------------------------------------------------------------
// GEMM1: h1[70000x128] = X[70000x1044] @ W1[1044x128]   (bf16 WMMA, f32 acc)
// block = 256 threads (8 waves).  Block tile: 128 rows x 128 cols.
// wave grid 4(M) x 2(N): each wave -> 32 rows x 64 cols = 2x4 WMMA tiles.
// Double-buffered LDS for the W1 chunk: one barrier per K-step; unguarded
// main loop for K<1024, guarded 20-wide tail outside the loop.
// ---------------------------------------------------------------------------
__device__ __forceinline__ v16bf load_bfrag_lds(const __bf16* base) {
  const uint4* q = reinterpret_cast<const uint4*>(base);
  BfPair p; p.lo = q[0]; p.hi = q[2];
  return __builtin_bit_cast(v16bf, p);
}

__device__ __forceinline__ v16bf load_afrag_fast(const float* __restrict__ row,
                                                 int k0, int koff) {
  v16bf a;
  const float* p = row + k0 + koff;            // 16B aligned (1044*4 % 16 == 0)
  float4 x0 = *(const float4*)(p);
  float4 x1 = *(const float4*)(p + 4);
  float4 x2 = *(const float4*)(p + 16);
  float4 x3 = *(const float4*)(p + 20);
  a[0]=(__bf16)x0.x;  a[1]=(__bf16)x0.y;  a[2]=(__bf16)x0.z;  a[3]=(__bf16)x0.w;
  a[4]=(__bf16)x1.x;  a[5]=(__bf16)x1.y;  a[6]=(__bf16)x1.z;  a[7]=(__bf16)x1.w;
  a[8]=(__bf16)x2.x;  a[9]=(__bf16)x2.y;  a[10]=(__bf16)x2.z; a[11]=(__bf16)x2.w;
  a[12]=(__bf16)x3.x; a[13]=(__bf16)x3.y; a[14]=(__bf16)x3.z; a[15]=(__bf16)x3.w;
  return a;
}

__device__ __forceinline__ v16bf load_afrag_tail(const float* __restrict__ row,
                                                 int k0, int koff) {
  v16bf a;
#pragma unroll
  for (int j = 0; j < 16; ++j) {
    int kk = k0 + koff + ((j < 8) ? j : j + 8);   // ISA 16-bit A 16x32 layout
    a[j] = (__bf16)((kk < kInCh) ? row[kk] : 0.0f);
  }
  return a;
}

__device__ __forceinline__ void stage_B_fast(__bf16* Bs, const float* __restrict__ W1,
                                             int k0, int bcol, int bkh) {
#pragma unroll
  for (int i = 0; i < 16; ++i) {
    float w = W1[(size_t)(k0 + bkh + i) * kFeat + bcol];
    Bs[bcol * kBStride + bkh + i] = (__bf16)w;
  }
}

__device__ __forceinline__ void stage_B_tail(__bf16* Bs, const float* __restrict__ W1,
                                             int k0, int bcol, int bkh) {
#pragma unroll
  for (int i = 0; i < 16; ++i) {
    int kk = k0 + bkh + i;
    float w = (kk < kInCh) ? W1[(size_t)kk * kFeat + bcol] : 0.0f;
    Bs[bcol * kBStride + bkh + i] = (__bf16)w;
  }
}

__device__ __forceinline__ void compute_step(const __bf16* Bs, v16bf a0, v16bf a1,
                                             v8f acc[2][4], int wn, int mrow, int half) {
  // Issue all 4 B-fragment LDS loads first (clause), then run the 8 WMMAs
  // back-to-back so the matrix pipe overlaps the remaining ds traffic.
  v16bf b[4];
#pragma unroll
  for (int nt = 0; nt < 4; ++nt) {
    int col = wn * 64 + nt * 16 + mrow;          // lane -> column (B layout)
    b[nt] = load_bfrag_lds(&Bs[col * kBStride + half * 8]);
  }
#pragma unroll
  for (int nt = 0; nt < 4; ++nt) {
    acc[0][nt] = __builtin_amdgcn_wmma_f32_16x16x32_bf16(
        false, a0, false, b[nt], (short)0, acc[0][nt], false, false);
    acc[1][nt] = __builtin_amdgcn_wmma_f32_16x16x32_bf16(
        false, a1, false, b[nt], (short)0, acc[1][nt], false, false);
  }
}

__global__ __launch_bounds__(256) void k_gemm1_wmma(const float* __restrict__ X,
                                                    const float* __restrict__ W1,
                                                    float* __restrict__ H1) {
  // B staged transposed in LDS (double buffered): Bs[buf][col][k], k-stride 40
  // bf16 (80B: 16B-aligned, bank-conflict-free b128 reads across 16 lanes).
  __shared__ __align__(16) __bf16 Bs[2][kFeat * kBStride];

  const int tid   = threadIdx.x;
  const int lane  = tid & 31;
  const int wave  = tid >> 5;
  const int wm    = wave & 3;           // M group: 32 rows each
  const int wn    = wave >> 2;          // N group: 64 cols each
  const int m0    = blockIdx.x * 128;
  const int half  = lane >> 4;          // 0: lanes 0-15, 1: lanes 16-31
  const int mrow  = lane & 15;
  const int koff  = half * 8;           // A-fragment K offset per ISA layout

  v8f acc[2][4] = {};                   // zero-init accumulators

  int rowA0 = m0 + wm * 32 + mrow;       if (rowA0 >= kNodes) rowA0 = kNodes - 1;
  int rowA1 = m0 + wm * 32 + 16 + mrow;  if (rowA1 >= kNodes) rowA1 = kNodes - 1;
  const float* pA0 = X + (size_t)rowA0 * kInCh;
  const float* pA1 = X + (size_t)rowA1 * kInCh;

  const int bcol = tid >> 1;            // 0..127
  const int bkh  = (tid & 1) * 16;      // 0 / 16

  // prologue: stage first chunk
  stage_B_fast(Bs[0], W1, 0, bcol, bkh);
  __syncthreads();

  int cur = 0;
  // main loop: unguarded A loads (compute K 0..1023, stage up to 1055; last stage guarded)
  for (int k0 = 0; k0 < kKMain; k0 += 32) {
    // stage next chunk into the other buffer while computing this one
    if (k0 + 32 < kKMain) {
      stage_B_fast(Bs[cur ^ 1], W1, k0 + 32, bcol, bkh);
    } else {
      stage_B_tail(Bs[cur ^ 1], W1, k0 + 32, bcol, bkh);  // chunk 1024..1055
    }
    v16bf a0 = load_afrag_fast(pA0, k0, koff);
    v16bf a1 = load_afrag_fast(pA1, k0, koff);
    __builtin_prefetch(pA0 + k0 + 64, 0, 3);   // global_prefetch_b8, near
    __builtin_prefetch(pA1 + k0 + 64, 0, 3);
    compute_step(Bs[cur], a0, a1, acc, wn, mrow, half);
    __syncthreads();
    cur ^= 1;
  }

  // tail: K = 1024..1043 (already staged, guarded A loads)
  {
    v16bf a0 = load_afrag_tail(pA0, kKMain, koff);
    v16bf a1 = load_afrag_tail(pA1, kKMain, koff);
    compute_step(Bs[cur], a0, a1, acc, wn, mrow, half);
  }

  // --- store C tiles (ISA 32-bit 16x16 C/D layout) ---
#pragma unroll
  for (int mt = 0; mt < 2; ++mt) {
#pragma unroll
    for (int nt = 0; nt < 4; ++nt) {
      int col = wn * 64 + nt * 16 + mrow;
#pragma unroll
      for (int r = 0; r < 8; ++r) {
        int row = m0 + wm * 32 + mt * 16 + (half ? 8 + r : r);
        if (row < kNodes) H1[(size_t)row * kFeat + col] = acc[mt][nt][r];
      }
    }
  }
}

// ---------------------------------------------------------------------------
// scatter layer 1: one wave per edge, float4 per lane (4 ch), 4 atomics
// ---------------------------------------------------------------------------
__global__ void k_scatter1(const long long* __restrict__ edges,
                           const float* __restrict__ dinv,
                           const float* __restrict__ H1,
                           float* __restrict__ agg) {
  long long gid = (long long)blockIdx.x * blockDim.x + threadIdx.x;
  int e   = (int)(gid >> 5);
  int sub = (int)(gid & 31);
  if (e >= kEdgesTot) return;
  int s, d;
  if (e < kEdges) {
    s = (int)edges[e];
    d = (int)edges[(size_t)kEdges + e];
  } else {
    s = d = e - kEdges;                 // self-loop
  }
  float nrm = dinv[s] * dinv[d];
  const float4 v = *(const float4*)(H1 + (size_t)s * kFeat + sub * 4);
  float* o = agg + (size_t)d * kFeat + sub * 4;
  atomicAdd(o + 0, v.x * nrm);
  atomicAdd(o + 1, v.y * nrm);
  atomicAdd(o + 2, v.z * nrm);
  atomicAdd(o + 3, v.w * nrm);
}

__global__ void k_bias_relu(float* __restrict__ x, const float* __restrict__ b1, int n) {
  int i = blockIdx.x * blockDim.x + threadIdx.x;
  if (i < n) {
    float v = x[i] + b1[i & (kFeat - 1)];
    x[i] = v > 0.0f ? v : 0.0f;
  }
}

// ---------------------------------------------------------------------------
// GEMM2 (128 -> 3): wave per node, shuffle reduction
// ---------------------------------------------------------------------------
__global__ __launch_bounds__(256) void k_gemm2(const float* __restrict__ X1,
                                               const float* __restrict__ W2,
                                               float* __restrict__ H2) {
  __shared__ float w2s[kFeat * kOut];
  for (int i = threadIdx.x; i < kFeat * kOut; i += blockDim.x) w2s[i] = W2[i];
  __syncthreads();
  int lane = threadIdx.x & 31;
  int node = blockIdx.x * 8 + (threadIdx.x >> 5);
  if (node >= kNodes) return;
  float s0 = 0.f, s1 = 0.f, s2 = 0.f;
#pragma unroll
  for (int kk = 0; kk < kFeat; kk += 32) {
    float v = X1[(size_t)node * kFeat + kk + lane];
    s0 += v * w2s[(kk + lane) * 3 + 0];
    s1 += v * w2s[(kk + lane) * 3 + 1];
    s2 += v * w2s[(kk + lane) * 3 + 2];
  }
#pragma unroll
  for (int off = 16; off > 0; off >>= 1) {
    s0 += __shfl_xor(s0, off, 32);
    s1 += __shfl_xor(s1, off, 32);
    s2 += __shfl_xor(s2, off, 32);
  }
  if (lane == 0) {
    H2[(size_t)node * 3 + 0] = s0;
    H2[(size_t)node * 3 + 1] = s1;
    H2[(size_t)node * 3 + 2] = s2;
  }
}

__global__ void k_init_out(float* out, const float* __restrict__ b2, int n) {
  int i = blockIdx.x * blockDim.x + threadIdx.x;
  if (i < n) out[i] = b2[i % 3];
}

__global__ void k_scatter2(const long long* __restrict__ edges,
                           const float* __restrict__ dinv,
                           const float* __restrict__ H2,
                           float* __restrict__ out) {
  int e = blockIdx.x * blockDim.x + threadIdx.x;
  if (e >= kEdgesTot) return;
  int s, d;
  if (e < kEdges) {
    s = (int)edges[e];
    d = (int)edges[(size_t)kEdges + e];
  } else {
    s = d = e - kEdges;
  }
  float nrm = dinv[s] * dinv[d];
#pragma unroll
  for (int c = 0; c < 3; ++c)
    atomicAdd(&out[(size_t)d * 3 + c], H2[(size_t)s * 3 + c] * nrm);
}

// ---------------------------------------------------------------------------
extern "C" void kernel_launch(void* const* d_in, const int* in_sizes, int n_in,
                              void* d_out, int out_size, void* d_ws, size_t ws_size,
                              hipStream_t stream) {
  const float*     X   = (const float*)d_in[0];
  const long long* edg = (const long long*)d_in[1];
  // d_in[2] edges2, d_in[3] edge_features: unused by the reference
  const float* W1 = (const float*)d_in[4];
  const float* b1 = (const float*)d_in[5];
  const float* W2 = (const float*)d_in[6];
  const float* b2 = (const float*)d_in[7];
  float* out = (float*)d_out;

  auto alignUp = [](size_t x) { return (x + 255) & ~(size_t)255; };
  char* ws = (char*)d_ws;
  size_t off = 0;
  float* deg  = (float*)(ws + off); off += alignUp((size_t)kNodes * 4);
  float* dinv = (float*)(ws + off); off += alignUp((size_t)kNodes * 4);
  float* h1   = (float*)(ws + off); off += alignUp((size_t)kNodes * kFeat * 4);
  float* agg1 = (float*)(ws + off); off += alignUp((size_t)kNodes * kFeat * 4);
  float* h2   = (float*)(ws + off); off += alignUp((size_t)kNodes * kOut * 4);
  (void)ws_size; (void)in_sizes; (void)n_in; (void)out_size;

  const int T = 256;
  // normalization
  k_init_deg <<<(kNodes + T - 1) / T, T, 0, stream>>>(deg, kNodes);
  k_accum_deg<<<(kEdges + T - 1) / T, T, 0, stream>>>(edg, deg, kEdges);
  k_dinv     <<<(kNodes + T - 1) / T, T, 0, stream>>>(deg, dinv, kNodes);
  // layer 1
  k_zero_f32 <<<(kNodes * kFeat + T - 1) / T, T, 0, stream>>>(agg1, kNodes * kFeat);
  k_gemm1_wmma<<<(kNodes + 127) / 128, T, 0, stream>>>(X, W1, h1);
  {
    long long total = (long long)kEdgesTot * 32;
    k_scatter1<<<(unsigned)((total + T - 1) / T), T, 0, stream>>>(edg, dinv, h1, agg1);
  }
  k_bias_relu<<<(kNodes * kFeat + T - 1) / T, T, 0, stream>>>(agg1, b1, kNodes * kFeat);
  // layer 2
  k_gemm2   <<<(kNodes + 7) / 8, T, 0, stream>>>(agg1, W2, h2);
  k_init_out<<<(kNodes * kOut + T - 1) / T, T, 0, stream>>>(out, b2, kNodes * kOut);
  k_scatter2<<<(kEdgesTot + T - 1) / T, T, 0, stream>>>(edg, dinv, h2, out);
}
